// Ch_Pos_GCL_83236466196758
// MI455X (gfx1250) — compile-verified, matrix operand order checked by reading, not run on hardware
//
#include <hip/hip_runtime.h>
#include <hip/hip_bf16.h>

// ---------------------------------------------------------------------------
// Types for CDNA5 WMMA (wave32): v_wmma_f32_16x16x32_bf16
// ---------------------------------------------------------------------------
typedef __attribute__((ext_vector_type(8)))  __bf16 v8bf;
typedef __attribute__((ext_vector_type(16))) __bf16 v16bf;
typedef __attribute__((ext_vector_type(8)))  float  v8f;

__device__ __forceinline__ __bf16 f2bf(float f) {
    union { float f; unsigned u; } x; x.f = f;
    unsigned r = x.u + 0x7FFFu + ((x.u >> 16) & 1u);   // round-to-nearest-even
    unsigned short hbits = (unsigned short)(r >> 16);
    return __builtin_bit_cast(__bf16, hbits);
}

__device__ __forceinline__ float siluf(float x) {
    return x / (1.f + __expf(-x));
}

// Build a 16x32 bf16 A/B fragment for one lane.
// Per ISA layout (16-bit A 16x32): lane holds k = k0+hi*8 .. +7  (elements 0..7)
// and k = k0+16+hi*8 .. +7 (elements 8..15).  Two 16-byte loads.
__device__ __forceinline__ v16bf ldfrag(const __bf16* p) {
    v8bf lo = *(const v8bf*)(p);
    v8bf hi = *(const v8bf*)(p + 16);
    v16bf r;
#pragma unroll
    for (int i = 0; i < 8; ++i) { r[i] = lo[i]; r[i + 8] = hi[i]; }
    return r;
}

__device__ __forceinline__ v8f wmma_bf16(v16bf a, v16bf b, v8f c) {
    return __builtin_amdgcn_wmma_f32_16x16x32_bf16(false, a, false, b,
                                                   (short)0, c, false, false);
}

// Dual-M-tile GEMM with software pipelining: C0/C1[16,16] += A0/A1[16,KDIM] x
// B[KDIM,16].  B fragment is loaded once and feeds two WMMAs (2x B reuse);
// next-k fragments are prefetched into fresh registers before the current
// WMMAs so global/ds loads stay in flight across the matrix ops.
template <int KDIM>
__device__ __forceinline__ void gemm2(const __bf16* a0_row, const __bf16* a1_row,
                                      const __bf16* b_row, int hi,
                                      v8f& acc0, v8f& acc1) {
    v16bf a0 = ldfrag(a0_row + hi * 8);
    v16bf a1 = ldfrag(a1_row + hi * 8);
    v16bf b  = ldfrag(b_row + hi * 8);
#pragma unroll
    for (int k0 = 32; k0 < KDIM; k0 += 32) {
        v16bf na0 = ldfrag(a0_row + k0 + hi * 8);
        v16bf na1 = ldfrag(a1_row + k0 + hi * 8);
        v16bf nb  = ldfrag(b_row + k0 + hi * 8);
        acc0 = wmma_bf16(a0, b, acc0);
        acc1 = wmma_bf16(a1, b, acc1);
        a0 = na0; a1 = na1; b = nb;
    }
    acc0 = wmma_bf16(a0, b, acc0);
    acc1 = wmma_bf16(a1, b, acc1);
}

// ---------------------------------------------------------------------------
// Prep: W[din,dout] f32  ->  Wt[dout, kpad] bf16 (zero-padded K), column-major
// by output so each B fragment is a contiguous 16B global load.
// ---------------------------------------------------------------------------
__global__ void transpose_w(const float* __restrict__ w, __bf16* __restrict__ wt,
                            int din, int dout, int kpad) {
    int i = blockIdx.x * blockDim.x + threadIdx.x;
    if (i >= dout * kpad) return;
    int n = i / kpad, k = i % kpad;
    float v = (k < din) ? w[(size_t)k * dout + n] : 0.f;
    wt[i] = f2bf(v);
}

// ---------------------------------------------------------------------------
// Main kernel: one block of 8 wave32 handles 32 edges (two 16-row M tiles).
// Wave w owns hidden columns [16w, 16w+16) for both tiles.
// ---------------------------------------------------------------------------
#define CS 520   // chem_in LDS row stride (bf16): 512 + 8 pad (16B-aligned, bank shift)
#define PS 40    // pos_in  LDS row stride (bf16): 32 + 8 pad
#define HS 136   // hidden  LDS row stride: 128 + 8 pad

__global__ __launch_bounds__(256)
void edge_mlp_kernel(const float* __restrict__ h, const float* __restrict__ coord,
                     const float* __restrict__ nvecs, const float* __restrict__ na,
                     const float* __restrict__ cb1, const float* __restrict__ cb2,
                     const float* __restrict__ pb1, const float* __restrict__ pb2,
                     const float* __restrict__ sb,  const float* __restrict__ aw,
                     const float* __restrict__ ab,
                     const int* __restrict__ edges, int E,
                     const __bf16* __restrict__ cw1t, const __bf16* __restrict__ cw2t,
                     const __bf16* __restrict__ pw1t, const __bf16* __restrict__ pw2t,
                     const __bf16* __restrict__ swt,
                     float* __restrict__ out, float* __restrict__ chem_out,
                     float* __restrict__ pos_out, float* __restrict__ cd_out) {
    __shared__ __align__(16) __bf16 sChem[32 * CS];   // gathered chem_in, bf16
    __shared__ __align__(16) __bf16 sPos [32 * PS];   // pos_in (padded 20->32)
    __shared__ __align__(16) __bf16 sHid1[32 * HS];   // silu(chem_in@cw1+cb1)
    __shared__ __align__(16) __bf16 sHidP[32 * HS];   // silu(pos_in @pw1+pb1)
    __shared__ __align__(16) __bf16 sHidC[32 * HS];   // chem (bf16, gate GEMM input)
    __shared__ __align__(16) float  sOut [32 * HS];   // pre-attention out (f32)
    __shared__ float sAtt[32];
    __shared__ int   sRow[32], sCol[32];

    const int t  = threadIdx.x;
    const int e0 = blockIdx.x * 32;

    if (t < 64) {                               // edge endpoints for this tile
        int e  = t & 31;
        int eg = e0 + e; if (eg >= E) eg = E - 1;
        if (t < 32) sRow[e] = edges[eg];
        else        sCol[e] = edges[(size_t)E + eg];
    }
    __syncthreads();

    // ---- gather chem_in = [h[row] | h[col] | na[row] | na[col]] as bf16 ----
#pragma unroll
    for (int i = 0; i < 64; ++i) {
        int idx = t + i * 256;                  // 32 rows x 512 cols
        int e = idx >> 9, c = idx & 511;
        int seg = c >> 7, off = c & 127;
        int node = (seg & 1) ? sCol[e] : sRow[e];
        const float* src = (seg < 2) ? h : na;
        sChem[e * CS + c] = f2bf(src[(size_t)node * 128 + off]);
    }

    // ---- pos_in (nprod[5] | radial[15] | 0-pad) + cd output ----
    if (t < 32) {
        int e = t, r = sRow[e], c = sCol[e];
        float dx = coord[r * 3 + 0] - coord[c * 3 + 0];
        float dy = coord[r * 3 + 1] - coord[c * 3 + 1];
        float dz = coord[r * 3 + 2] - coord[c * 3 + 2];
        float inv = 1.f / (sqrtf(dx * dx + dy * dy + dz * dz) + 1e-8f);
        dx *= inv; dy *= inv; dz *= inv;
        int eg = e0 + e;
        if (eg < E) {
            cd_out[(size_t)eg * 3 + 0] = dx;
            cd_out[(size_t)eg * 3 + 1] = dy;
            cd_out[(size_t)eg * 3 + 2] = dz;
        }
        float ir = dx * dx + dy * dy + dz * dz;
#pragma unroll
        for (int k = 0; k < 5; ++k) {
            const float* a = nvecs + ((size_t)r * 5 + k) * 3;
            const float* b = nvecs + ((size_t)c * 5 + k) * 3;
            sPos[e * PS + k] = f2bf(a[0] * b[0] + a[1] * b[1] + a[2] * b[2]);
        }
        float sc = -0.5f;                       // SCALES[s] = -0.5 / 2.25^s
#pragma unroll
        for (int s = 0; s < 15; ++s) {
            sPos[e * PS + 5 + s] = f2bf(__expf(ir * sc));
            sc *= (1.f / 2.25f);
        }
#pragma unroll
        for (int k = 20; k < 32; ++k) sPos[e * PS + k] = f2bf(0.f);
    }
    __syncthreads();

    const int lane = t & 31;
    const int w    = t >> 5;
    const int n    = lane & 15;     // A row / B column / D column within tile
    const int hi   = lane >> 4;
    const int col  = w * 16 + n;    // global hidden column owned by this lane

    // ---- stage A: chem1 (K=512, 32 WMMA) and pos1 (K=32, 2 WMMA) ----
    {
        v8f c0 = {}, c1 = {};
        gemm2<512>(&sChem[n * CS], &sChem[(16 + n) * CS],
                   cw1t + (size_t)col * 512, hi, c0, c1);
        v8f p0 = {}, p1 = {};
        gemm2<32>(&sPos[n * PS], &sPos[(16 + n) * PS],
                  pw1t + (size_t)col * 32, hi, p0, p1);
        float bc = cb1[col], bp = pb1[col];
#pragma unroll
        for (int v = 0; v < 8; ++v) {
            int m0 = hi * 8 + v, m1 = 16 + m0;  // D layout: lane holds rows hi*8..+7
            sHid1[m0 * HS + col] = f2bf(siluf(c0[v] + bc));
            sHid1[m1 * HS + col] = f2bf(siluf(c1[v] + bc));
            sHidP[m0 * HS + col] = f2bf(siluf(p0[v] + bp));
            sHidP[m1 * HS + col] = f2bf(siluf(p1[v] + bp));
        }
    }
    __syncthreads();

    // ---- stage B: chem2 and pos2 (K=128, 8+8 WMMA) ----
    v8f posv0, posv1;
    {
        v8f c0 = {}, c1 = {};
        gemm2<128>(&sHid1[n * HS], &sHid1[(16 + n) * HS],
                   cw2t + (size_t)col * 128, hi, c0, c1);
        v8f p0 = {}, p1 = {};
        gemm2<128>(&sHidP[n * HS], &sHidP[(16 + n) * HS],
                   pw2t + (size_t)col * 128, hi, p0, p1);
        float bc = cb2[col], bp = pb2[col];
#pragma unroll
        for (int v = 0; v < 8; ++v) {
            int m0 = hi * 8 + v, m1 = 16 + m0;
            float cv0 = siluf(c0[v] + bc), cv1 = siluf(c1[v] + bc);
            float pv0 = siluf(p0[v] + bp), pv1 = siluf(p1[v] + bp);
            posv0[v] = pv0; posv1[v] = pv1;
            sHidC[m0 * HS + col] = f2bf(cv0);
            sHidC[m1 * HS + col] = f2bf(cv1);
            int eg0 = e0 + m0, eg1 = e0 + m1;
            if (eg0 < E) {
                chem_out[(size_t)eg0 * 128 + col] = cv0;
                pos_out [(size_t)eg0 * 128 + col] = pv0;
            }
            if (eg1 < E) {
                chem_out[(size_t)eg1 * 128 + col] = cv1;
                pos_out [(size_t)eg1 * 128 + col] = pv1;
            }
        }
    }
    __syncthreads();

    // ---- stage C: gate GEMM (K=128, 8 WMMA), out = silu(chem@sw+sb)*pos ----
    {
        v8f g0 = {}, g1 = {};
        gemm2<128>(&sHidC[n * HS], &sHidC[(16 + n) * HS],
                   swt + (size_t)col * 128, hi, g0, g1);
        float b = sb[col];
#pragma unroll
        for (int v = 0; v < 8; ++v) {
            int m0 = hi * 8 + v, m1 = 16 + m0;
            sOut[m0 * HS + col] = siluf(g0[v] + b) * posv0[v];
            sOut[m1 * HS + col] = siluf(g1[v] + b) * posv1[v];
        }
    }
    __syncthreads();

    // ---- attention: att[e] = sigmoid(out[e,:] . aw + ab) ----
    if (t < 32) {
        float s = ab[0];
        for (int c = 0; c < 128; ++c) s += sOut[t * HS + c] * aw[c];
        sAtt[t] = 1.f / (1.f + __expf(-s));
    }
    __syncthreads();

    // ---- final scaled store ----
#pragma unroll
    for (int i = 0; i < 16; ++i) {
        int idx = t + i * 256;                  // 32 x 128
        int e = idx >> 7, c = idx & 127;
        int eg = e0 + e;
        if (eg < E) out[(size_t)eg * 128 + c] = sOut[e * HS + c] * sAtt[e];
    }
}

// ---------------------------------------------------------------------------
// Launch.  d_ws layout (16B-aligned bf16 transposed weights):
//   cw1t [128][512] @ 0        (131072 B)
//   cw2t [128][128] @ 131072   ( 32768 B)
//   pw1t [128][ 32] @ 163840   (  8192 B)
//   pw2t [128][128] @ 172032   ( 32768 B)
//   swt  [128][128] @ 204800   ( 32768 B)   -> total 237568 B
// ---------------------------------------------------------------------------
extern "C" void kernel_launch(void* const* d_in, const int* in_sizes, int n_in,
                              void* d_out, int out_size, void* d_ws, size_t ws_size,
                              hipStream_t stream) {
    const float* h     = (const float*)d_in[0];
    const float* coord = (const float*)d_in[1];
    const float* nvecs = (const float*)d_in[2];
    const float* na    = (const float*)d_in[3];
    const float* cw1   = (const float*)d_in[4];
    const float* cb1   = (const float*)d_in[5];
    const float* cw2   = (const float*)d_in[6];
    const float* cb2   = (const float*)d_in[7];
    const float* pw1   = (const float*)d_in[8];
    const float* pb1   = (const float*)d_in[9];
    const float* pw2   = (const float*)d_in[10];
    const float* pb2   = (const float*)d_in[11];
    const float* sw    = (const float*)d_in[12];
    const float* sb    = (const float*)d_in[13];
    const float* aw    = (const float*)d_in[14];
    const float* ab    = (const float*)d_in[15];
    const int*   edges = (const int*)d_in[16];
    const int E = in_sizes[16] / 2;

    char* ws = (char*)d_ws;
    __bf16* cw1t = (__bf16*)(ws + 0);
    __bf16* cw2t = (__bf16*)(ws + 131072);
    __bf16* pw1t = (__bf16*)(ws + 163840);
    __bf16* pw2t = (__bf16*)(ws + 172032);
    __bf16* swt  = (__bf16*)(ws + 204800);

    transpose_w<<<(128 * 512 + 255) / 256, 256, 0, stream>>>(cw1, cw1t, 512, 128, 512);
    transpose_w<<<(128 * 128 + 255) / 256, 256, 0, stream>>>(cw2, cw2t, 128, 128, 128);
    transpose_w<<<(128 *  32 + 255) / 256, 256, 0, stream>>>(pw1, pw1t,  20, 128,  32);
    transpose_w<<<(128 * 128 + 255) / 256, 256, 0, stream>>>(pw2, pw2t, 128, 128, 128);
    transpose_w<<<(128 * 128 + 255) / 256, 256, 0, stream>>>(sw,  swt,  128, 128, 128);

    float* out  = (float*)d_out;
    float* chem = out  + (size_t)E * 128;
    float* pos  = chem + (size_t)E * 128;
    float* cd   = pos  + (size_t)E * 128;

    const int blocks = (E + 31) / 32;
    edge_mlp_kernel<<<blocks, 256, 0, stream>>>(h, coord, nvecs, na,
                                                cb1, cb2, pb1, pb2, sb, aw, ab,
                                                edges, E,
                                                cw1t, cw2t, pw1t, pw2t, swt,
                                                out, chem, pos, cd);
}
